// Set2SetReadout_55705725829534
// MI455X (gfx1250) — matrix operand order repaired and checked by reading, not compiled, not run on hardware
//
#include <hip/hip_runtime.h>
#include <hip/hip_bf16.h>
#include <math.h>

// ---------------- problem constants (match reference) ----------------
#define N_GRAPHS 4096
#define HIDDEN   256
#define OUT_DIM  256
#define STEPS    6
#define N_NODES  524288
#define GATE_N   (4 * HIDDEN)   // 1024
#define XK       (2 * HIDDEN)   // 512

typedef __attribute__((ext_vector_type(16))) __bf16 v16bf;
typedef __attribute__((ext_vector_type(8)))  float  v8f;

union Frag { v16bf v; unsigned u[8]; };

// ---------------------------------------------------------------------
// Segment boundaries: batch_indices is sorted; seg[g] = lower_bound(g).
// ---------------------------------------------------------------------
__global__ void seg_bounds_kernel(const int* __restrict__ idx, int* __restrict__ seg, int n) {
    int g = blockIdx.x * blockDim.x + threadIdx.x;
    if (g > N_GRAPHS) return;
    if (g == N_GRAPHS) { seg[N_GRAPHS] = n; return; }
    int lo = 0, hi = n;
    while (lo < hi) { int mid = (lo + hi) >> 1; if (idx[mid] < g) lo = mid + 1; else hi = mid; }
    seg[g] = lo;
}

__global__ void zero_f32_kernel(float* __restrict__ p, int n) {
    int i = blockIdx.x * blockDim.x + threadIdx.x;
    if (i < n) p[i] = 0.f;
}

// ---------------------------------------------------------------------
// Weight prep: Wg[n][k] (bf16, 1024x512) = k<256 ? W_ih[n][k]+W_hh[n][k]
//                                        : W_ih[n][k];  bias fold too.
// ---------------------------------------------------------------------
__global__ void prep_gate_w_kernel(const float* __restrict__ W_ih, const float* __restrict__ W_hh,
                                   __bf16* __restrict__ Wg) {
    int i = blockIdx.x * blockDim.x + threadIdx.x;      // 1024*512
    if (i >= GATE_N * XK) return;
    int n = i >> 9, k = i & (XK - 1);
    float v = W_ih[n * XK + k];
    if (k < HIDDEN) v += W_hh[n * HIDDEN + k];
    Wg[i] = (__bf16)v;
}

__global__ void prep_bias_kernel(const float* __restrict__ b_ih, const float* __restrict__ b_hh,
                                 float* __restrict__ bg) {
    int i = blockIdx.x * blockDim.x + threadIdx.x;
    if (i < GATE_N) bg[i] = b_ih[i] + b_hh[i];
}

__global__ void f32_to_bf16_kernel(const float* __restrict__ s, __bf16* __restrict__ d, int n) {
    int i = blockIdx.x * blockDim.x + threadIdx.x;
    if (i < n) d[i] = (__bf16)s[i];
}

// xb = bf16([h | r])  (4096 x 512)
__global__ void build_xb_kernel(const float* __restrict__ h, const float* __restrict__ r,
                                __bf16* __restrict__ xb) {
    int i = blockIdx.x * blockDim.x + threadIdx.x;      // 4096*512
    if (i >= N_GRAPHS * XK) return;
    int g = i >> 9, k = i & (XK - 1);
    float v = (k < HIDDEN) ? h[g * HIDDEN + k] : r[g * HIDDEN + (k - HIDDEN)];
    xb[i] = (__bf16)v;
}

// ---------------------------------------------------------------------
// Attention readout, ONE streaming pass (online softmax).
// One 256-thread block per graph; 8 waves stride the graph's node range.
// ---------------------------------------------------------------------
__global__ __launch_bounds__(256) void attn_readout_kernel(
        const float* __restrict__ x,     // node_embeddings (N x 256)
        const int*   __restrict__ seg,   // N_GRAPHS+1
        const float* __restrict__ h,     // (G x 256)
        float*       __restrict__ r) {   // (G x 256)
    const int g = blockIdx.x;
    __shared__ float hs[HIDDEN];
    __shared__ float m_sh[8], s_sh[8];
    __shared__ float racc_sh[8][HIDDEN];

    const int t = threadIdx.x;
    hs[t] = h[g * HIDDEN + t];
    __syncthreads();

    const int beg = seg[g], end = seg[g + 1];
    const int wave = t >> 5, lane = t & 31;

    float hv[8];
    #pragma unroll
    for (int j = 0; j < 4; ++j) { hv[j] = hs[4 * lane + j]; hv[4 + j] = hs[128 + 4 * lane + j]; }

    float m = -INFINITY, s = 0.f;
    float racc[8] = {0.f, 0.f, 0.f, 0.f, 0.f, 0.f, 0.f, 0.f};

    for (int n = beg + wave; n < end; n += 8) {
        const float* row = x + (size_t)n * HIDDEN;
        __builtin_prefetch(row + 8 * HIDDEN + 4 * lane, 0, 1);   // global_prefetch_b8
        const float4 xa = *(const float4*)(row + 4 * lane);
        const float4 xc = *(const float4*)(row + 128 + 4 * lane);
        float part = xa.x * hv[0] + xa.y * hv[1] + xa.z * hv[2] + xa.w * hv[3]
                   + xc.x * hv[4] + xc.y * hv[5] + xc.z * hv[6] + xc.w * hv[7];
        #pragma unroll
        for (int off = 16; off; off >>= 1) part += __shfl_xor(part, off, 32);
        const float e  = part;
        const float mn = fmaxf(m, e);
        const float sc = __expf(m - mn);     // 0 when m==-inf
        const float p  = __expf(e - mn);
        s = s * sc + p;
        racc[0] = racc[0] * sc + p * xa.x;  racc[1] = racc[1] * sc + p * xa.y;
        racc[2] = racc[2] * sc + p * xa.z;  racc[3] = racc[3] * sc + p * xa.w;
        racc[4] = racc[4] * sc + p * xc.x;  racc[5] = racc[5] * sc + p * xc.y;
        racc[6] = racc[6] * sc + p * xc.z;  racc[7] = racc[7] * sc + p * xc.w;
        m = mn;
    }

    if (lane == 0) { m_sh[wave] = m; s_sh[wave] = s; }
    #pragma unroll
    for (int j = 0; j < 4; ++j) {
        racc_sh[wave][4 * lane + j]       = racc[j];
        racc_sh[wave][128 + 4 * lane + j] = racc[4 + j];
    }
    __syncthreads();

    // combine 8 waves; thread t owns output element t
    float M = -INFINITY;
    #pragma unroll
    for (int w = 0; w < 8; ++w) M = fmaxf(M, m_sh[w]);
    float S = 0.f, R = 0.f;
    if (M > -INFINITY) {
        #pragma unroll
        for (int w = 0; w < 8; ++w) {
            const float sc = __expf(m_sh[w] - M);
            S += s_sh[w] * sc;
            R += racc_sh[w][t] * sc;
        }
        r[g * HIDDEN + t] = R / (S + 1e-16f);
    } else {
        r[g * HIDDEN + t] = 0.f;   // empty graph
    }
}

// ---------------------------------------------------------------------
// bf16 WMMA GEMM:  out(MxN) = A(MxK,bf16) * W(NxK,bf16)^T + bias
// K is a template parameter -> the chunk loop fully unrolls, and the
// two fragment buffers (a0/b0, a1/b1) are selected by the COMPILE-TIME
// parity of the chunk index: no runtime-indexed arrays, so all fragments
// stay in VGPRs (round-2 version spilled to scratch). Loads for chunk
// ch+1 are issued before the 4 v_wmma of chunk ch.
// Fragment index math follows CDNA5 ISA 7.12.2 (wave32 layouts).
// ---------------------------------------------------------------------
template<int K>
__global__ __launch_bounds__(256) void gemm_bf16_kernel(
        const __bf16* __restrict__ A,
        const __bf16* __restrict__ W,
        const float*  __restrict__ bias,
        float* __restrict__ outF, __bf16* __restrict__ outB,
        int M, int N, int relu) {
    const int wavesPerN = N >> 6;                       // N/64
    const int gw   = (blockIdx.x * blockDim.x + threadIdx.x) >> 5;
    const int lane = threadIdx.x & 31;
    const int m0 = (gw / wavesPerN) << 4;
    const int n0 = (gw % wavesPerN) << 6;
    if (m0 >= M) return;

    const int half = lane >> 4;       // 0: lanes 0-15, 1: lanes 16-31
    const int lm   = lane & 15;

    // A 16x32 bf16: lane<16 -> K {0..7,16..23}; lane>=16 -> K {8..15,24..31}
    const __bf16* arow0 = A + (size_t)(m0 + lm) * K + half * 8;
    // B 32x16 bf16: lane<16 -> col N=lm, K {0..7,16..23}; lane>=16 -> K {8..15,24..31}
    const __bf16* wrow0 = W + (size_t)(n0 + lm) * K + half * 8;
    const size_t  wstep = (size_t)16 * K;               // next 16-column tile

    auto loadA = [&](Frag& a, int k0) {
        const __bf16* p = arow0 + k0;
        #pragma unroll
        for (int j = 0; j < 4; ++j) {
            a.u[j]     = *(const unsigned*)(p + 2 * j);
            a.u[4 + j] = *(const unsigned*)(p + 16 + 2 * j);
        }
    };
    auto loadB = [&](Frag (&b)[4], int k0) {
        #pragma unroll
        for (int tn = 0; tn < 4; ++tn) {
            const __bf16* p = wrow0 + (size_t)tn * wstep + k0;
            #pragma unroll
            for (int j = 0; j < 4; ++j) { b[tn].u[j] = *(const unsigned*)(p + 2 * j); b[tn].u[4 + j] = *(const unsigned*)(p + 16 + 2 * j); }
        }
    };

    v8f acc[4] = {};
    auto wmma4 = [&](Frag& a, Frag (&b)[4]) {
        #pragma unroll
        for (int tn = 0; tn < 4; ++tn)
            acc[tn] = __builtin_amdgcn_wmma_f32_16x16x32_bf16(
                          false, a.v, false, b[tn].v, (short)0, acc[tn], false, false);
    };

    constexpr int NCH = K / 32;       // 16 (K=512) or 8 (K=256)
    Frag a0, a1;
    Frag b0[4], b1[4];
    loadA(a0, 0);
    loadB(b0, 0);

    #pragma unroll
    for (int ch = 0; ch < NCH; ++ch) {
        if ((ch & 1) == 0) {
            if (ch + 1 < NCH) { loadA(a1, (ch + 1) * 32); loadB(b1, (ch + 1) * 32); }
            wmma4(a0, b0);
        } else {
            if (ch + 1 < NCH) { loadA(a0, (ch + 1) * 32); loadB(b0, (ch + 1) * 32); }
            wmma4(a1, b1);
        }
    }

    // C/D: VGPR v -> row m0 + v + half*8, col n0 + tn*16 + lm
    #pragma unroll
    for (int tn = 0; tn < 4; ++tn) {
        const int n = n0 + tn * 16 + lm;
        const float bb = bias ? bias[n] : 0.f;
        #pragma unroll
        for (int v = 0; v < 8; ++v) {
            const int mrow = m0 + v + half * 8;
            float val = acc[tn][v] + bb;
            if (relu) val = fmaxf(val, 0.f);
            if (outF) outF[(size_t)mrow * N + n] = val;
            else      outB[(size_t)mrow * N + n] = (__bf16)val;
        }
    }
}

// ---------------------------------------------------------------------
// LSTM pointwise: c' = sig(f)*c + sig(i)*tanh(g);  h' = sig(o)*tanh(c')
// ---------------------------------------------------------------------
__device__ __forceinline__ float sigf(float x) { return 1.f / (1.f + __expf(-x)); }

__global__ void lstm_ew_kernel(const float* __restrict__ gates,
                               float* __restrict__ h, float* __restrict__ c) {
    int i = blockIdx.x * blockDim.x + threadIdx.x;      // 4096*256
    if (i >= N_GRAPHS * HIDDEN) return;
    const int g = i >> 8, u = i & (HIDDEN - 1);
    const float* row = gates + (size_t)g * GATE_N;
    const float ig = row[u];
    const float fg = row[HIDDEN + u];
    const float gg = row[2 * HIDDEN + u];
    const float og = row[3 * HIDDEN + u];
    const float cn = sigf(fg) * c[i] + sigf(ig) * tanhf(gg);
    c[i] = cn;
    h[i] = sigf(og) * tanhf(cn);
}

// ---------------------------------------------------------------------
extern "C" void kernel_launch(void* const* d_in, const int* in_sizes, int n_in,
                              void* d_out, int out_size, void* d_ws, size_t ws_size,
                              hipStream_t stream) {
    const float* node = (const float*)d_in[0];
    const int*   bidx = (const int*)d_in[1];
    const float* W_ih = (const float*)d_in[2];
    const float* W_hh = (const float*)d_in[3];
    const float* b_ih = (const float*)d_in[4];
    const float* b_hh = (const float*)d_in[5];
    const float* W1   = (const float*)d_in[6];
    const float* b1   = (const float*)d_in[7];
    const float* W2   = (const float*)d_in[8];
    const float* b2   = (const float*)d_in[9];
    float* out = (float*)d_out;

    // --- workspace carve (256B aligned) ---
    char* wsp = (char*)d_ws;
    auto carve = [&](size_t bytes) { char* p = wsp; wsp += (bytes + 255) & ~(size_t)255; return p; };
    int*    seg    = (int*)   carve(sizeof(int) * (N_GRAPHS + 1));
    float*  h      = (float*) carve(sizeof(float) * N_GRAPHS * HIDDEN);
    float*  c      = (float*) carve(sizeof(float) * N_GRAPHS * HIDDEN);
    float*  r      = (float*) carve(sizeof(float) * N_GRAPHS * HIDDEN);
    float*  gates  = (float*) carve(sizeof(float) * N_GRAPHS * GATE_N);
    __bf16* xb     = (__bf16*)carve(sizeof(__bf16) * N_GRAPHS * XK);
    __bf16* hdnb   = (__bf16*)carve(sizeof(__bf16) * N_GRAPHS * HIDDEN);
    __bf16* Wgbf   = (__bf16*)carve(sizeof(__bf16) * GATE_N * XK);
    float*  biasg  = (float*) carve(sizeof(float) * GATE_N);
    __bf16* W1bf   = (__bf16*)carve(sizeof(__bf16) * HIDDEN * XK);
    __bf16* W2bf   = (__bf16*)carve(sizeof(__bf16) * OUT_DIM * HIDDEN);

    const int TB = 256;
    auto blocks = [](int n, int tb) { return (n + tb - 1) / tb; };

    // init + prep
    zero_f32_kernel<<<blocks(N_GRAPHS * HIDDEN, TB), TB, 0, stream>>>(h, N_GRAPHS * HIDDEN);
    zero_f32_kernel<<<blocks(N_GRAPHS * HIDDEN, TB), TB, 0, stream>>>(c, N_GRAPHS * HIDDEN);
    seg_bounds_kernel<<<blocks(N_GRAPHS + 1, TB), TB, 0, stream>>>(bidx, seg, N_NODES);
    prep_gate_w_kernel<<<blocks(GATE_N * XK, TB), TB, 0, stream>>>(W_ih, W_hh, Wgbf);
    prep_bias_kernel<<<blocks(GATE_N, TB), TB, 0, stream>>>(b_ih, b_hh, biasg);
    f32_to_bf16_kernel<<<blocks(HIDDEN * XK, TB), TB, 0, stream>>>(W1, W1bf, HIDDEN * XK);
    f32_to_bf16_kernel<<<blocks(OUT_DIM * HIDDEN, TB), TB, 0, stream>>>(W2, W2bf, OUT_DIM * HIDDEN);

    // GEMM launch geometry: one wave per 16x64 tile
    auto gemm_blocks = [&](int M, int N) { return blocks((M / 16) * (N / 64) * 32, TB); };

    for (int step = 0; step < STEPS; ++step) {
        attn_readout_kernel<<<N_GRAPHS, TB, 0, stream>>>(node, seg, h, r);
        build_xb_kernel<<<blocks(N_GRAPHS * XK, TB), TB, 0, stream>>>(h, r, xb);
        gemm_bf16_kernel<XK><<<gemm_blocks(N_GRAPHS, GATE_N), TB, 0, stream>>>(
            xb, Wgbf, biasg, gates, (__bf16*)nullptr, N_GRAPHS, GATE_N, 0);
        lstm_ew_kernel<<<blocks(N_GRAPHS * HIDDEN, TB), TB, 0, stream>>>(gates, h, c);
    }

    // final readout MLP:  hdn = relu([h|r] @ W1^T + b1) ; out = hdn @ W2^T + b2
    build_xb_kernel<<<blocks(N_GRAPHS * XK, TB), TB, 0, stream>>>(h, r, xb);
    gemm_bf16_kernel<XK><<<gemm_blocks(N_GRAPHS, HIDDEN), TB, 0, stream>>>(
        xb, W1bf, b1, (float*)nullptr, hdnb, N_GRAPHS, HIDDEN, 1);
    gemm_bf16_kernel<HIDDEN><<<gemm_blocks(N_GRAPHS, OUT_DIM), TB, 0, stream>>>(
        hdnb, W2bf, b2, out, (__bf16*)nullptr, N_GRAPHS, OUT_DIM, 0);
}
